// TwoStageBiMambaEncoder_34772055229102
// MI455X (gfx1250) — matrix operand order repaired and checked by reading, not compiled, not run on hardware
//
#include <hip/hip_runtime.h>
#include <stddef.h>

// ---------------- types ----------------
typedef __attribute__((ext_vector_type(16))) __bf16 v16bf;
typedef __attribute__((ext_vector_type(8)))  float  v8f;
typedef __attribute__((ext_vector_type(4)))  unsigned int v4u;
typedef __attribute__((ext_vector_type(8)))  int    v8i_;
typedef __attribute__((ext_vector_type(4)))  int    v4i_;

#if defined(__HIP_DEVICE_COMPILE__) && \
    __has_builtin(__builtin_amdgcn_tensor_load_to_lds) && \
    __has_builtin(__builtin_amdgcn_s_wait_tensorcnt)
#define USE_TDM 1
#else
#define USE_TDM 0
#endif

__device__ __forceinline__ float sigmoidf_(float x) { return 1.f / (1.f + expf(-x)); }
__device__ __forceinline__ float siluf_(float x)    { return x * sigmoidf_(x); }
__device__ __forceinline__ float geluf_(float x)    { return 0.5f * x * (1.f + erff(x * 0.70710678118654752f)); }
__device__ __forceinline__ float softplusf_(float x){ return (x > 30.f) ? x : log1pf(expf(x)); }

#if USE_TDM
// byte offset of a __shared__ object inside the workgroup's LDS allocation
__device__ __forceinline__ unsigned lds_off_(const void* p) {
    return (unsigned)(unsigned long long)
        (__attribute__((address_space(3))) const char*)p;
}
// Issue a TDM DMA of one 32(x) x 16(y) fp32 tile, row stride ld (elements),
// from global byte-address ga into LDS byte-offset lds.  D# per ISA ch.8.
// This toolchain's builtin takes 6 args (g0, g1, g2, g3, g4, cpol).
__device__ __forceinline__ void tdm_tile_load_(unsigned long long ga, unsigned lds, int ld) {
    v4u g0;
    g0[0] = 1u;                                   // count=1 (valid descriptor)
    g0[1] = lds;                                  // lds_addr (bytes)
    g0[2] = (unsigned)(ga & 0xFFFFFFFFu);         // global_addr[31:0]
    g0[3] = (unsigned)((ga >> 32) & 0x01FFFFFFu)  // global_addr[56:32]
          | (2u << 30);                           // type=2 ("image")
    v8i_ g1;
    g1[0] = 0x20000;                              // wg_mask=0, data_size=2 (4B)
    g1[1] = 0;                                    // no atomic barrier; dim0[15:0]=0
    g1[2] = (int)((1u << 30) >> 16);              // dim0=2^30 (no clip); dim1 lo16=0
    g1[3] = (int)(((1u << 30) >> 16) | (32u << 16)); // dim1=2^30; tile_dim0=32
    g1[4] = 16;                                   // tile_dim1=16, tile_dim2=0
    g1[5] = ld;                                   // tensor_dim0_stride (elements)
    g1[6] = 0;                                    // stride hi, dim1_stride lo
    g1[7] = 0;
    __builtin_amdgcn_tensor_load_to_lds(g0, g1, (v4i_)0, (v4i_)0, (v8i_)0, 0);
}
#endif

// =====================================================================
// Generic GEMM: C[M,N] = epilogue(A[M,K] * Bw[N,K]^T + bias[N]) + addsrc
// One wave per 16x16 C tile; K-loop steps 32 through
// v_wmma_f32_16x16x32_bf16 (fp32 accumulate).
// Fast path (EDGE=false): tiles staged by the Tensor Data Mover into
// ping-pong LDS buffers, synchronized with TENSORcnt (no barriers needed,
// single wave per block).  Fallback: wide vector loads.
// ep: 0=none, 1=gelu, 2=softplus
// =====================================================================
template <bool EDGE>
__global__ __launch_bounds__(32)
void gemm_bf16_wmma(const float* __restrict__ A, int lda,
                    const float* __restrict__ Bw, int ldb,
                    float* __restrict__ C, int ldc,
                    int M, int N, int K,
                    const float* __restrict__ bias,
                    const float* __restrict__ addsrc,
                    int ep)
{
    const int lane = threadIdx.x;
    const int row0 = blockIdx.y * 16;   // M tile
    const int col0 = blockIdx.x * 16;   // N tile

    union { v8f v; float e[8]; } acc;
    #pragma unroll
    for (int i = 0; i < 8; ++i) acc.e[i] = 0.f;

    // fragment gather coordinates (ISA 7.12.2 layouts)
    const int mrow = lane & 15;
    const int kbA  = (lane < 16) ? 0 : 8;
    const int kbB  = (lane < 16) ? 0 : 8;

    if (!EDGE) {
#if USE_TDM
        // ---------------- TDM-staged fp32 tiles, ping-pong ----------------
        __shared__ alignas(16) float sAf[2][16][32];
        __shared__ alignas(16) float sBf[2][16][32];
        const unsigned loA[2] = { lds_off_(&sAf[0][0][0]), lds_off_(&sAf[1][0][0]) };
        const unsigned loB[2] = { lds_off_(&sBf[0][0][0]), lds_off_(&sBf[1][0][0]) };
        const unsigned long long gaA = (unsigned long long)(uintptr_t)A
                                     + 4ull * ((unsigned long long)row0 * (unsigned)lda);
        const unsigned long long gaB = (unsigned long long)(uintptr_t)Bw
                                     + 4ull * ((unsigned long long)col0 * (unsigned)ldb);

        tdm_tile_load_(gaA, loA[0], lda);
        tdm_tile_load_(gaB, loB[0], ldb);

        int p = 0;
        for (int k0 = 0; k0 < K; k0 += 32, p ^= 1) {
            const bool more = (k0 + 32 < K);
            if (more) {
                tdm_tile_load_(gaA + 4ull * (unsigned)(k0 + 32), loA[p ^ 1], lda);
                tdm_tile_load_(gaB + 4ull * (unsigned)(k0 + 32), loB[p ^ 1], ldb);
                __builtin_amdgcn_s_wait_tensorcnt(2);  // current pair complete (in-order)
            } else {
                __builtin_amdgcn_s_wait_tensorcnt(0);
            }
            // gather fp32 runs from LDS, convert to bf16 fragments
            union { float4 q[4]; float f[16]; } ra, rb;
            ra.q[0] = *(const float4*)&sAf[p][mrow][kbA + 0];
            ra.q[1] = *(const float4*)&sAf[p][mrow][kbA + 4];
            ra.q[2] = *(const float4*)&sAf[p][mrow][16 + kbA + 0];
            ra.q[3] = *(const float4*)&sAf[p][mrow][16 + kbA + 4];
            rb.q[0] = *(const float4*)&sBf[p][mrow][kbB + 0];
            rb.q[1] = *(const float4*)&sBf[p][mrow][kbB + 4];
            rb.q[2] = *(const float4*)&sBf[p][mrow][16 + kbB + 0];
            rb.q[3] = *(const float4*)&sBf[p][mrow][16 + kbB + 4];
            union { v16bf v; __bf16 e[16]; } fa, fb;
            #pragma unroll
            for (int j = 0; j < 16; ++j) { fa.e[j] = (__bf16)ra.f[j]; fb.e[j] = (__bf16)rb.f[j]; }
            acc.v = __builtin_amdgcn_wmma_f32_16x16x32_bf16(
                false, fa.v, false, fb.v, (short)0, acc.v, false, false);
        }
#else
        // ---------------- synchronous wide-load fallback ----------------
        __shared__ alignas(16) __bf16 sA[16][32];
        __shared__ alignas(16) __bf16 sB[16][32];
        const int lr = lane >> 1;
        const int lk = (lane & 1) * 16;
        const float* Arow = A  + (size_t)(row0 + lr) * lda + lk;
        const float* Brow = Bw + (size_t)(col0 + lr) * ldb + lk;
        uint4* dA = (uint4*)&sA[lr][lk];
        uint4* dB = (uint4*)&sB[lr][lk];
        for (int k0 = 0; k0 < K; k0 += 32) {
            const float4* pa = (const float4*)(Arow + k0);
            const float4* pb = (const float4*)(Brow + k0);
            union { float4 f4[4]; float f[16]; } va, vb;
            va.f4[0] = pa[0]; va.f4[1] = pa[1]; va.f4[2] = pa[2]; va.f4[3] = pa[3];
            vb.f4[0] = pb[0]; vb.f4[1] = pb[1]; vb.f4[2] = pb[2]; vb.f4[3] = pb[3];
            if (k0 + 32 < K) {
                __builtin_prefetch(Arow + k0 + 32, 0, 1);
                __builtin_prefetch(Brow + k0 + 32, 0, 1);
            }
            union { __bf16 h[16]; uint4 q[2]; } ha, hb;
            #pragma unroll
            for (int j = 0; j < 16; ++j) { ha.h[j] = (__bf16)va.f[j]; hb.h[j] = (__bf16)vb.f[j]; }
            dA[0] = ha.q[0]; dA[1] = ha.q[1];
            dB[0] = hb.q[0]; dB[1] = hb.q[1];
            __syncthreads();
            union { v16bf v; __bf16 e[16]; } fa, fb;
            #pragma unroll
            for (int j = 0; j < 16; ++j) {
                int q = j >> 1, o = j & 1;
                int ka = (q < 4) ? (kbA + 2 * q + o) : (16 + kbA + 2 * (q - 4) + o);
                int kb = (q < 4) ? (kbB + 2 * q + o) : (16 + kbB + 2 * (q - 4) + o);
                fa.e[j] = sA[mrow][ka];
                fb.e[j] = sB[mrow][kb];
            }
            acc.v = __builtin_amdgcn_wmma_f32_16x16x32_bf16(
                false, fa.v, false, fb.v, (short)0, acc.v, false, false);
            __syncthreads();
        }
#endif
    } else {
        // ---------------- edge path: branchless clamped scalar staging ----------------
        __shared__ alignas(16) __bf16 sA[16][32];
        __shared__ alignas(16) __bf16 sB[16][32];
        const int lr = lane >> 1;
        const int lk = (lane & 1) * 16;
        const bool aok = (row0 + lr) < M;
        const bool bok = (col0 + lr) < N;
        const int  arc = aok ? (row0 + lr) : (M - 1);
        const int  brc = bok ? (col0 + lr) : (N - 1);
        for (int k0 = 0; k0 < K; k0 += 32) {
            #pragma unroll
            for (int j = 0; j < 16; ++j) {
                int gk  = k0 + lk + j;
                int gkc = (gk < K) ? gk : (K - 1);
                float av = A [(size_t)arc * lda + gkc];
                float bv = Bw[(size_t)brc * ldb + gkc];
                av = (aok && gk < K) ? av : 0.f;
                bv = (bok && gk < K) ? bv : 0.f;
                sA[lr][lk + j] = (__bf16)av;
                sB[lr][lk + j] = (__bf16)bv;
            }
            __syncthreads();
            union { v16bf v; __bf16 e[16]; } fa, fb;
            #pragma unroll
            for (int j = 0; j < 16; ++j) {
                int q = j >> 1, o = j & 1;
                int ka = (q < 4) ? (kbA + 2 * q + o) : (16 + kbA + 2 * (q - 4) + o);
                int kb = (q < 4) ? (kbB + 2 * q + o) : (16 + kbB + 2 * (q - 4) + o);
                fa.e[j] = sA[mrow][ka];
                fb.e[j] = sB[mrow][kb];
            }
            acc.v = __builtin_amdgcn_wmma_f32_16x16x32_bf16(
                false, fa.v, false, fb.v, (short)0, acc.v, false, false);
            __syncthreads();
        }
    }

    // ---- epilogue + store (C layout: VGPR p -> M = p (+8 for lanes 16-31))
    const int cn = col0 + (lane & 15);
    const int mb = row0 + ((lane < 16) ? 0 : 8);
    #pragma unroll
    for (int p = 0; p < 8; ++p) {
        int cm = mb + p;
        if (EDGE && (cm >= M || cn >= N)) continue;
        float v = acc.e[p];
        if (bias)    v += bias[cn];
        if (ep == 1) v = geluf_(v);
        else if (ep == 2) v = softplusf_(v);
        if (addsrc)  v += addsrc[(size_t)cm * ldc + cn];
        C[(size_t)cm * ldc + cn] = v;
    }
}

// =====================================================================
// Depthwise causal conv (D_CONV=4) + SiLU.  dir=+1 causal, dir=-1 the
// flipped-sequence conv in natural index order (anti-causal).
// =====================================================================
__global__ void conv_silu_k(const float* __restrict__ xz, float* __restrict__ xc,
                            const float* __restrict__ cw, const float* __restrict__ cb,
                            int Bsz, int L, int di, int dir)
{
    int idx = blockIdx.x * blockDim.x + threadIdx.x;
    if (idx >= Bsz * di) return;
    int b = idx / di, e = idx % di;
    float w0 = cw[e * 4 + 0], w1 = cw[e * 4 + 1], w2 = cw[e * 4 + 2], w3 = cw[e * 4 + 3];
    float bias = cb[e];
    const float* xi = xz + (size_t)b * L * (2 * (size_t)di) + e;
    float*       oc = xc + (size_t)b * L * (size_t)di + e;
    const size_t st = 2 * (size_t)di;
    for (int l = 0; l < L; ++l) {
        float s = bias;
        if (dir > 0) {
            if (l - 3 >= 0) s += w0 * xi[(size_t)(l - 3) * st];
            if (l - 2 >= 0) s += w1 * xi[(size_t)(l - 2) * st];
            if (l - 1 >= 0) s += w2 * xi[(size_t)(l - 1) * st];
            s += w3 * xi[(size_t)l * st];
        } else {
            s += w3 * xi[(size_t)l * st];
            if (l + 1 < L) s += w2 * xi[(size_t)(l + 1) * st];
            if (l + 2 < L) s += w1 * xi[(size_t)(l + 2) * st];
            if (l + 3 < L) s += w0 * xi[(size_t)(l + 3) * st];
        }
        oc[(size_t)l * di] = siluf_(s);
    }
}

// =====================================================================
// Selective-scan per (b, channel): 8-state recurrence over L, fused with
// y = (h.C + xc*D) * silu(z).  dir=-1 runs L-1..0.
// =====================================================================
__global__ void scan_k(const float* __restrict__ xz,
                       const float* __restrict__ xc,
                       const float* __restrict__ dt,
                       const float* __restrict__ xdbl,
                       const float* __restrict__ A_log,
                       const float* __restrict__ Dp,
                       float* __restrict__ ys,
                       int Bsz, int L, int di, int dt_rank, int dir)
{
    int idx = blockIdx.x * blockDim.x + threadIdx.x;
    if (idx >= Bsz * di) return;
    int b = idx / di, e = idx % di;

    float A[8];
    #pragma unroll
    for (int n = 0; n < 8; ++n) A[n] = -expf(A_log[(size_t)e * 8 + n]);
    float Dv = Dp[e];
    float h[8];
    #pragma unroll
    for (int n = 0; n < 8; ++n) h[n] = 0.f;

    const int xdw = dt_rank + 16;
    for (int t = 0; t < L; ++t) {
        int l = (dir > 0) ? t : (L - 1 - t);
        size_t base = (size_t)b * L + l;
        float dtv = dt[base * di + e];
        float xcv = xc[base * di + e];
        const float* xd = xdbl + base * xdw + dt_rank;
        float y = 0.f;
        #pragma unroll
        for (int n = 0; n < 8; ++n) {
            float dA = expf(dtv * A[n]);
            h[n] = dA * h[n] + dtv * xd[n] * xcv;
            y += h[n] * xd[8 + n];
        }
        float zv = xz[base * (2 * (size_t)di) + di + e];
        ys[base * di + e] = (y + xcv * Dv) * siluf_(zv);
    }
}

// =====================================================================
// LayerNorm: one wave per row, shuffle reduction (wave32). In-place safe.
// =====================================================================
__global__ __launch_bounds__(32)
void layernorm_k(const float* __restrict__ x, float* __restrict__ out,
                 const float* __restrict__ g, const float* __restrict__ b, int D)
{
    int row = blockIdx.x, lane = threadIdx.x;
    const float* xr = x + (size_t)row * D;
    float s = 0.f, s2 = 0.f;
    for (int c = lane; c < D; c += 32) { float v = xr[c]; s += v; s2 += v * v; }
    #pragma unroll
    for (int o = 16; o > 0; o >>= 1) { s += __shfl_xor(s, o, 32); s2 += __shfl_xor(s2, o, 32); }
    float mu = s / (float)D;
    float var = s2 / (float)D - mu * mu;
    float r = rsqrtf(var + 1e-5f);
    float* orow = out + (size_t)row * D;
    for (int c = lane; c < D; c += 32) orow[c] = (xr[c] - mu) * r * g[c] + b[c];
}

// =====================================================================
// Transposed token projection: (16,14,512) -> (16,512,14) @ (14,64) + b
// =====================================================================
__global__ void tokproj_k(const float* __restrict__ tin, const float* __restrict__ W,
                          const float* __restrict__ bias, float* __restrict__ out)
{
    int idx = blockIdx.x * blockDim.x + threadIdx.x;
    if (idx >= 16 * 512 * 64) return;
    int s = idx & 63;
    int d = (idx >> 6) & 511;
    int b = idx >> 15;
    float acc = bias[s];
    #pragma unroll
    for (int v = 0; v < 14; ++v)
        acc += tin[((size_t)b * 14 + v) * 512 + d] * W[v * 64 + s];
    out[idx] = acc;
}

// ====================== host-side parameter plumbing ======================
struct MambaP { const float *W_in, *conv_w, *conv_b, *W_x, *W_dt, *b_dt, *A_log, *D, *W_out; };
struct LayerP { MambaP mf, mr; const float *ln1_g, *ln1_b, *W1, *b1, *W2, *b2, *ln2_g, *ln2_b; };
struct EncP   { LayerP L[4]; const float *ln_g, *ln_b; };

struct Cursor { void* const* din; int i; };
static inline const float* nx(Cursor& c) { return (const float*)c.din[c.i++]; }

static void mamba_ins (Cursor& c, MambaP& m){ m.W_in=nx(c); m.conv_w=nx(c); m.conv_b=nx(c); m.W_x=nx(c); m.W_dt=nx(c); m.b_dt=nx(c); m.A_log=nx(c); m.D=nx(c); m.W_out=nx(c); }
static void mamba_sort(Cursor& c, MambaP& m){ m.A_log=nx(c); m.D=nx(c); m.W_dt=nx(c); m.W_in=nx(c); m.W_out=nx(c); m.W_x=nx(c); m.b_dt=nx(c); m.conv_b=nx(c); m.conv_w=nx(c); }
static void layer_ins (Cursor& c, LayerP& l){ mamba_ins(c,l.mf); mamba_ins(c,l.mr); l.ln1_g=nx(c); l.ln1_b=nx(c); l.W1=nx(c); l.b1=nx(c); l.W2=nx(c); l.b2=nx(c); l.ln2_g=nx(c); l.ln2_b=nx(c); }
static void layer_sort(Cursor& c, LayerP& l){ l.W1=nx(c); l.W2=nx(c); l.b1=nx(c); l.b2=nx(c); l.ln1_b=nx(c); l.ln1_g=nx(c); l.ln2_b=nx(c); l.ln2_g=nx(c); mamba_sort(c,l.mf); mamba_sort(c,l.mr); }
static void enc_ins   (Cursor& c, EncP& e){ for (int i=0;i<4;++i) layer_ins (c,e.L[i]); e.ln_g=nx(c); e.ln_b=nx(c); }
static void enc_sort  (Cursor& c, EncP& e){ for (int i=0;i<4;++i) layer_sort(c,e.L[i]); e.ln_b=nx(c); e.ln_g=nx(c); }

static inline void launch_gemm(hipStream_t st, const float* A, int lda, const float* Bw, int ldb,
                               float* C, int ldc, int M, int N, int K,
                               const float* bias, const float* add, int ep)
{
    dim3 g((N + 15) / 16, (M + 15) / 16);
    bool edge = (M & 15) || (N & 15) || (K & 31);
    if (edge)
        gemm_bf16_wmma<true ><<<g, dim3(32), 0, st>>>(A, lda, Bw, ldb, C, ldc, M, N, K, bias, add, ep);
    else
        gemm_bf16_wmma<false><<<g, dim3(32), 0, st>>>(A, lda, Bw, ldb, C, ldc, M, N, K, bias, add, ep);
}

static void run_encoder(hipStream_t st, float* xbuf, const EncP& ep,
                        int Bsz, int L, int D, int d_inner, int dt_rank, int d_ff,
                        float* xz, float* xc, float* xdbl, float* dt, float* ys,
                        float* newx, float* xln, float* ffmid)
{
    const int rows = Bsz * L;
    const int xdw  = dt_rank + 16;
    const int cthreads = Bsz * d_inner;
    const int cblocks  = (cthreads + 255) / 256;

    for (int li = 0; li < 4; ++li) {
        const LayerP& lp = ep.L[li];
        for (int pass = 0; pass < 2; ++pass) {
            const MambaP& m = (pass == 0) ? lp.mf : lp.mr;
            const int dir = (pass == 0) ? 1 : -1;
            launch_gemm(st, xbuf, D, m.W_in, D, xz, 2 * d_inner, rows, 2 * d_inner, D, nullptr, nullptr, 0);
            conv_silu_k<<<cblocks, 256, 0, st>>>(xz, xc, m.conv_w, m.conv_b, Bsz, L, d_inner, dir);
            launch_gemm(st, xc, d_inner, m.W_x, d_inner, xdbl, xdw, rows, xdw, d_inner, nullptr, nullptr, 0);
            launch_gemm(st, xdbl, xdw, m.W_dt, dt_rank, dt, d_inner, rows, d_inner, dt_rank, m.b_dt, nullptr, 2);
            scan_k<<<cblocks, 256, 0, st>>>(xz, xc, dt, xdbl, m.A_log, m.D, ys, Bsz, L, d_inner, dt_rank, dir);
            launch_gemm(st, ys, d_inner, m.W_out, d_inner, newx, D, rows, D, d_inner,
                        nullptr, pass ? newx : nullptr, 0);
        }
        layernorm_k<<<rows, 32, 0, st>>>(newx, xln, lp.ln1_g, lp.ln1_b, D);
        launch_gemm(st, xln, D, lp.W1, D, ffmid, d_ff, rows, d_ff, D, lp.b1, nullptr, 1);
        launch_gemm(st, ffmid, d_ff, lp.W2, d_ff, xbuf, D, rows, D, d_ff, lp.b2, xln, 0);
        layernorm_k<<<rows, 32, 0, st>>>(xbuf, xbuf, lp.ln2_g, lp.ln2_b, D);
    }
    layernorm_k<<<rows, 32, 0, st>>>(xbuf, xbuf, ep.ln_g, ep.ln_b, D);
}

// ====================== workspace layout (floats) ======================
#define WS_XBUF   ((size_t)0)
#define WS_XZ     (WS_XBUF  +  524288)
#define WS_XC     (WS_XZ    + 4194304)
#define WS_XDBL   (WS_XC    + 2097152)
#define WS_DT     (WS_XDBL  +  163840)
#define WS_YS     (WS_DT    + 2097152)
#define WS_NEWX   (WS_YS    + 2097152)
#define WS_XLN    (WS_NEWX  +  524288)
#define WS_FFMID  (WS_XLN   +  524288)

extern "C" void kernel_launch(void* const* d_in, const int* in_sizes, int n_in,
                              void* d_out, int out_size, void* d_ws, size_t ws_size,
                              hipStream_t stream)
{
    (void)n_in; (void)ws_size;
    const float* enc_in = (const float*)d_in[0];

    EncP tp, sp; const float *tokW, *tokb;
    Cursor c{d_in, 1};
    if (in_sizes[1] == 4096 * 512) {
        enc_ins(c, tp); enc_ins(c, sp); tokW = nx(c); tokb = nx(c);
    } else {
        enc_sort(c, sp); enc_sort(c, tp); tokW = nx(c); tokb = nx(c);
    }

    float* w     = (float*)d_ws;
    float* xbuf  = w + WS_XBUF;
    float* xz    = w + WS_XZ;
    float* xc    = w + WS_XC;
    float* xdbl  = w + WS_XDBL;
    float* dt    = w + WS_DT;
    float* ys    = w + WS_YS;
    float* newx  = w + WS_NEWX;
    float* xln   = w + WS_XLN;
    float* ffmid = w + WS_FFMID;

    (void)hipMemcpyAsync(xbuf, enc_in, (size_t)16 * 14 * 512 * sizeof(float),
                         hipMemcpyDeviceToDevice, stream);
    run_encoder(stream, xbuf, tp, 16, 14, 512, 2048, 32, 2048,
                xz, xc, xdbl, dt, ys, newx, xln, ffmid);

    tokproj_k<<<(16 * 512 * 64 + 255) / 256, 256, 0, stream>>>(xbuf, tokW, tokb, xln);
    (void)hipMemcpyAsync(xbuf, xln, (size_t)16 * 512 * 64 * sizeof(float),
                         hipMemcpyDeviceToDevice, stream);

    run_encoder(stream, xbuf, sp, 16, 512, 64, 256, 4, 2048,
                xz, xc, xdbl, dt, ys, newx, xln, ffmid);

    (void)hipMemcpyAsync(d_out, xbuf, (size_t)out_size * sizeof(float),
                         hipMemcpyDeviceToDevice, stream);
}
